// MultiHeadAttention_37469294690835
// MI455X (gfx1250) — matrix operand order repaired and checked by reading, not compiled, not run on hardware
//
#include <hip/hip_runtime.h>

// MI455X (gfx1250, wave32) implementation.
// Roofline: 537MB attn_weights write dominates (~23us @ 23.3TB/s); 43 GFLOP total
// compute done in bf16 WMMA (f32 accum). K pre-scaled by 1/sqrt(HD) so logits are
// a pure chained-WMMA accumulation; V stored transposed for contiguous B-operand
// loads; softmax uses online max/sum (pass 1) + recompute (pass 2) so logits never
// touch HBM; prob tiles are written to HBM as 128B-contiguous float4 rows via LDS.

#define B_   2
#define S_   2048
#define E_   1024
#define H_   16
#define HD_  32
#define ER_  512
#define SCALE_ 0.17677669529663687f   // 1/sqrt(32)

typedef __attribute__((ext_vector_type(16))) __bf16 v16bf;
typedef __attribute__((ext_vector_type(8)))  __bf16 v8bf;
typedef __attribute__((ext_vector_type(4)))  __bf16 v4bf;
typedef __attribute__((ext_vector_type(8)))  float  v8f;

__device__ __forceinline__ v8f wmma_bf16(v16bf a, v16bf b, v8f c) {
  return __builtin_amdgcn_wmma_f32_16x16x32_bf16(false, a, false, b, (short)0, c,
                                                 false, false);
}

// ---------------------------------------------------------------- rel -> bf16
__global__ void k_cvt_bf16(const float* __restrict__ src, __bf16* __restrict__ dst,
                           int n) {
  int i = blockIdx.x * 256 + threadIdx.x;
  if (i < n) dst[i] = (__bf16)src[i];
}

// ------------------------------------------------------------- QKV projection
// grid: (B*S/16, 24); block 128 (4 waves). by/8 selects Q/K/V; (by%8)*64 = n0.
// Q,K stored (b,h,s,d) bf16; K pre-scaled by 1/sqrt(HD); V stored (b,h,d,s).
__global__ void __launch_bounds__(128)
k_qkv(const float* __restrict__ x,
      const float* __restrict__ Wq, const float* __restrict__ bq,
      const float* __restrict__ Wk, const float* __restrict__ bk,
      const float* __restrict__ Wv, const float* __restrict__ bv,
      __bf16* __restrict__ qs, __bf16* __restrict__ ksc, __bf16* __restrict__ vsT)
{
  __shared__ __align__(64) __bf16 aT[16][32];
  __shared__ __align__(64) __bf16 bT[64][32];

  const int t = threadIdx.x;
  const int lane = t & 31, wave = t >> 5;
  const int m0 = blockIdx.x * 16;
  const int mat = blockIdx.y >> 3;
  const int n0 = (blockIdx.y & 7) * 64;
  const float* W; const float* bias;
  if (mat == 0)      { W = Wq; bias = bq; }
  else if (mat == 1) { W = Wk; bias = bk; }
  else               { W = Wv; bias = bv; }

  v8f acc = {};
  for (int kc = 0; kc < E_; kc += 32) {
    __syncthreads();
    { int idx = t * 4, r = idx >> 5, c = idx & 31;
      float4 xv = *(const float4*)(x + (size_t)(m0 + r) * E_ + kc + c);
      aT[r][c] = (__bf16)xv.x; aT[r][c+1] = (__bf16)xv.y;
      aT[r][c+2] = (__bf16)xv.z; aT[r][c+3] = (__bf16)xv.w; }
    #pragma unroll
    for (int u = 0; u < 4; ++u) {
      int idx = (t + u * 128) * 4, n = idx >> 5, c = idx & 31;
      float4 wv = *(const float4*)(W + (size_t)(n0 + n) * E_ + kc + c);
      bT[n][c] = (__bf16)wv.x; bT[n][c+1] = (__bf16)wv.y;
      bT[n][c+2] = (__bf16)wv.z; bT[n][c+3] = (__bf16)wv.w; }
    __syncthreads();
    int rA = lane & 15, kb = (lane < 16) ? 0 : 16;
    v8bf lo = *(const v8bf*)&aT[rA][kb];
    v8bf hi = *(const v8bf*)&aT[rA][kb + 8];
    v16bf a;
    #pragma unroll
    for (int i = 0; i < 8; ++i) { a[i] = lo[i]; a[i + 8] = hi[i]; }
    int nl = (wave << 4) + (lane & 15);
    v16bf brg = *(const v16bf*)&bT[nl][(lane < 16) ? 0 : 16];
    acc = wmma_bf16(a, brg, acc);
  }
  int colER = n0 + (wave << 4) + (lane & 15);
  int hh = colER >> 5, d = colER & 31;
  float bsv = bias[colER];
  int rbm = m0 + ((lane >> 4) << 3);
  #pragma unroll
  for (int e = 0; e < 8; ++e) {
    int m = rbm + e, b = m >> 11, s = m & (S_ - 1);
    float v = acc[e] + bsv;
    if (mat == 1) v *= SCALE_;
    if (mat == 2)
      vsT[(((size_t)b * H_ + hh) * HD_ + d) * S_ + s] = (__bf16)v;
    else {
      __bf16* dst = (mat == 0) ? qs : ksc;
      dst[(((size_t)b * H_ + hh) * S_ + s) * HD_ + d] = (__bf16)v;
    }
  }
}

// ------------------------------------------------------------------ attention
// grid: (S/16, B*H); block 128 (4 waves). One 16-row strip per block.
// Pass 1: chained WMMA logits (K + rel), online row max/sum-exp.
// Pass 2: recompute, stage normalized probs in LDS, write them out as coalesced
//         float4 rows (the 537MB roofline stream), accumulate P@V via WMMA.
__global__ void __launch_bounds__(128)
k_attn(const __bf16* __restrict__ qs, const __bf16* __restrict__ ksc,
       const __bf16* __restrict__ vsT, const __bf16* __restrict__ rels,
       float* __restrict__ attn, __bf16* __restrict__ av)
{
  __shared__ float pmaxS[4][32][8];
  __shared__ float psumS[4][32][8];
  __shared__ float rmaxS[16], rinvS[16];
  __shared__ __align__(64) float pbuf[4][16][32];
  __shared__ float red[4][16][32];

  const int t = threadIdx.x, lane = t & 31, wave = t >> 5;
  const int s0 = blockIdx.x * 16;
  const int bh = blockIdx.y;
  const int b = bh >> 4, h = bh & 15;

  // Q fragment (16x32, full head-dim) — identical in every wave of the block.
  const __bf16* qbase = qs + ((size_t)bh * S_ + s0) * HD_;
  const int rA = lane & 15, kbA = (lane < 16) ? 0 : 16;
  v8bf qlo = *(const v8bf*)(qbase + rA * HD_ + kbA);
  v8bf qhi = *(const v8bf*)(qbase + rA * HD_ + kbA + 8);
  v16bf qf;
  #pragma unroll
  for (int i = 0; i < 8; ++i) { qf[i] = qlo[i]; qf[i + 8] = qhi[i]; }

  const int kb2 = (lane < 16) ? 0 : 16;
  const int c   = lane & 15;
  const int rb  = (lane >> 4) << 3;

  // ---- pass 1: online softmax stats over 2048 columns ----
  float mrun[8], srun[8];
  #pragma unroll
  for (int e = 0; e < 8; ++e) { mrun[e] = -__builtin_inff(); srun[e] = 0.f; }
  for (int jt = wave; jt < 128; jt += 4) {
    int key = (jt << 4) + c;
    v16bf kf = *(const v16bf*)(ksc + ((size_t)bh * S_ + key) * HD_ + kb2);
    v16bf rf = *(const v16bf*)(rels + ((size_t)h * S_ + key) * HD_ + kb2);
    v8f acc = {};
    acc = wmma_bf16(qf, kf, acc);   // q . (k*scale)^T
    acc = wmma_bf16(qf, rf, acc);   // + q . rel^T
    #pragma unroll
    for (int e = 0; e < 8; ++e) {
      float v = acc[e];
      if (v > mrun[e]) {            // rare path: 1 exp to rescale running sum
        srun[e] = srun[e] * __expf(mrun[e] - v) + 1.f;
        mrun[e] = v;
      } else {                      // common path: 1 exp
        srun[e] += __expf(v - mrun[e]);
      }
    }
  }
  #pragma unroll
  for (int e = 0; e < 8; ++e) { pmaxS[wave][lane][e] = mrun[e];
                                psumS[wave][lane][e] = srun[e]; }
  __syncthreads();
  if (t < 16) {
    int row = t, lh = (row >> 3) << 4, e = row & 7;
    float M = -__builtin_inff(), Ssum = 0.f;
    for (int w2 = 0; w2 < 4; ++w2)
      for (int li = 0; li < 16; ++li) {
        float m_ = pmaxS[w2][lh + li][e], s_ = psumS[w2][lh + li][e];
        float nm = fmaxf(M, m_);
        Ssum = Ssum * __expf(M - nm) + s_ * __expf(m_ - nm);
        M = nm;
      }
    rmaxS[row] = M; rinvS[row] = 1.f / Ssum;
  }
  __syncthreads();

  // ---- pass 2: normalized probs out + P@V ----
  v8f apv0 = {}, apv1 = {};
  const __bf16* vb = vsT + (size_t)bh * HD_ * S_;
  for (int ch = wave; ch < 64; ch += 4) {        // uniform 16 iters per wave
    #pragma unroll
    for (int hf = 0; hf < 2; ++hf) {
      int col0 = (ch << 5) + (hf << 4);
      int key = col0 + c;
      v16bf kf = *(const v16bf*)(ksc + ((size_t)bh * S_ + key) * HD_ + kb2);
      v16bf rf = *(const v16bf*)(rels + ((size_t)h * S_ + key) * HD_ + kb2);
      v8f acc = {};
      acc = wmma_bf16(qf, kf, acc);
      acc = wmma_bf16(qf, rf, acc);
      #pragma unroll
      for (int e = 0; e < 8; ++e) {
        int row = rb + e;
        pbuf[wave][row][(hf << 4) + c] =
            __expf(acc[e] - rmaxS[row]) * rinvS[row];
      }
    }
    __syncthreads();                 // D-layout -> A-layout + coalesced writeout
    // 16x32 prob tile -> HBM as float4 rows (128B contiguous per row)
    {
      float* abase = attn + ((size_t)bh * S_ + s0) * S_ + (ch << 5);
      #pragma unroll
      for (int k = 0; k < 4; ++k) {
        int idx = (k << 7) + (lane << 2);        // 0..508, step 4
        int row = idx >> 5, col = idx & 31;
        float4 pv = *(const float4*)&pbuf[wave][row][col];
        *(float4*)(abase + (size_t)row * S_ + col) = pv;
      }
    }
    v16bf pf;
    #pragma unroll
    for (int i = 0; i < 8; ++i) {
      pf[i]     = (__bf16)pbuf[wave][rA][kbA + i];
      pf[i + 8] = (__bf16)pbuf[wave][rA][kbA + 8 + i];
    }
    v16bf v0 = *(const v16bf*)(vb + (size_t)c * S_ + (ch << 5) + kb2);
    v16bf v1 = *(const v16bf*)(vb + (size_t)(16 + c) * S_ + (ch << 5) + kb2);
    apv0 = wmma_bf16(pf, v0, apv0);
    apv1 = wmma_bf16(pf, v1, apv1);
    __syncthreads();
  }

  #pragma unroll
  for (int e = 0; e < 8; ++e) {
    red[wave][rb + e][c]      = apv0[e];
    red[wave][rb + e][16 + c] = apv1[e];
  }
  __syncthreads();
  for (int i = t; i < 512; i += 128) {
    int row = i >> 5, d = i & 31;
    float sum = red[0][row][d] + red[1][row][d] + red[2][row][d] + red[3][row][d];
    av[((size_t)b * S_ + s0 + row) * ER_ + h * HD_ + d] = (__bf16)sum;
  }
}

// ------------------------------------------------------------ out projection
// grid: (B*S/16, E/64); block 128. out = av @ Wo^T + bo.
__global__ void __launch_bounds__(128)
k_outproj(const __bf16* __restrict__ av, const float* __restrict__ Wo,
          const float* __restrict__ bo, float* __restrict__ out)
{
  __shared__ __align__(64) __bf16 aT[16][32];
  __shared__ __align__(64) __bf16 bT[64][32];
  const int t = threadIdx.x, lane = t & 31, wave = t >> 5;
  const int m0 = blockIdx.x * 16;
  const int n0 = blockIdx.y * 64;
  v8f acc = {};
  for (int kc = 0; kc < ER_; kc += 32) {
    __syncthreads();
    { int idx = t * 4, r = idx >> 5, cc = idx & 31;
      *(v4bf*)&aT[r][cc] = *(const v4bf*)(av + (size_t)(m0 + r) * ER_ + kc + cc); }
    #pragma unroll
    for (int u = 0; u < 4; ++u) {
      int idx = (t + u * 128) * 4, n = idx >> 5, cc = idx & 31;
      float4 wv = *(const float4*)(Wo + (size_t)(n0 + n) * ER_ + kc + cc);
      bT[n][cc] = (__bf16)wv.x; bT[n][cc+1] = (__bf16)wv.y;
      bT[n][cc+2] = (__bf16)wv.z; bT[n][cc+3] = (__bf16)wv.w; }
    __syncthreads();
    int rA = lane & 15, kb = (lane < 16) ? 0 : 16;
    v8bf lo = *(const v8bf*)&aT[rA][kb];
    v8bf hi = *(const v8bf*)&aT[rA][kb + 8];
    v16bf a;
    #pragma unroll
    for (int i = 0; i < 8; ++i) { a[i] = lo[i]; a[i + 8] = hi[i]; }
    int nl = (wave << 4) + (lane & 15);
    v16bf brg = *(const v16bf*)&bT[nl][(lane < 16) ? 0 : 16];
    acc = wmma_bf16(a, brg, acc);
  }
  int nG = n0 + (wave << 4) + (lane & 15);
  float bsv = bo[nG];
  int rbm = m0 + ((lane >> 4) << 3);
  #pragma unroll
  for (int e = 0; e < 8; ++e)
    out[(size_t)(rbm + e) * E_ + nG] = acc[e] + bsv;
}

// ------------------------------------------------------------------- launcher
extern "C" void kernel_launch(void* const* d_in, const int* in_sizes, int n_in,
                              void* d_out, int out_size, void* d_ws, size_t ws_size,
                              hipStream_t stream) {
  (void)in_sizes; (void)n_in; (void)out_size; (void)ws_size;
  const float* x   = (const float*)d_in[0];
  const float* Wq  = (const float*)d_in[1];
  const float* bq  = (const float*)d_in[2];
  const float* Wk  = (const float*)d_in[3];
  const float* bk  = (const float*)d_in[4];
  const float* Wv  = (const float*)d_in[5];
  const float* bv  = (const float*)d_in[6];
  const float* Wo  = (const float*)d_in[7];
  const float* bo  = (const float*)d_in[8];
  const float* rel = (const float*)d_in[9];
  float* out  = (float*)d_out;
  float* attn = out + (size_t)B_ * S_ * E_;

  const size_t QKV_EL = (size_t)B_ * H_ * S_ * HD_;   // 2,097,152 bf16 each
  const size_t REL_EL = (size_t)H_ * S_ * HD_;        // 1,048,576 bf16
  __bf16* qs   = (__bf16*)d_ws;
  __bf16* ksc  = qs + QKV_EL;
  __bf16* vsT  = ksc + QKV_EL;
  __bf16* rels = vsT + QKV_EL;
  __bf16* av   = rels + REL_EL;                       // total ~18 MB of ws

  k_cvt_bf16<<<(int)((REL_EL + 255) / 256), 256, 0, stream>>>(rel, rels,
                                                              (int)REL_EL);
  k_qkv<<<dim3((B_ * S_) / 16, 24), 128, 0, stream>>>(x, Wq, bq, Wk, bk, Wv, bv,
                                                      qs, ksc, vsT);
  k_attn<<<dim3(S_ / 16, B_ * H_), 128, 0, stream>>>(qs, ksc, vsT, rels, attn, av);
  k_outproj<<<dim3((B_ * S_) / 16, E_ / 64), 128, 0, stream>>>(av, Wo, bo, out);
}
